// AttackLinkPredictor_66898410602822
// MI455X (gfx1250) — compile-verified, matrix operand order checked by reading, not run on hardware
//
#include <hip/hip_runtime.h>
#include <hip/hip_bf16.h>

typedef __attribute__((ext_vector_type(16))) _Float16 v16h;
typedef __attribute__((ext_vector_type(8)))  float    v8f;
typedef __attribute__((ext_vector_type(4)))  int      v4i;

#define D 128
// LDS row strides padded so consecutive rows advance 4 banks (68 / 132 DWORDs
// == 4 mod 64): a b128 fragment read is bank-conflict-free across 16 lanes,
// and 272 B / 528 B row pitch keeps 16-byte alignment for ds_load_b128.
#define LDK128 136
#define LDK256 264

#if defined(__has_builtin)
#if __has_builtin(__builtin_amdgcn_global_load_async_to_lds_b128)
#define HAVE_ASYNC_LDS 1
#endif
#if __has_builtin(__builtin_amdgcn_s_wait_asynccnt)
#define HAVE_WAIT_ASYNC 1
#endif
#endif

// builtin takes (global int4*, lds int4*, imm offset, imm cpol)
typedef __attribute__((address_space(1))) v4i gv4i;
typedef __attribute__((address_space(3))) v4i lv4i;

// 16-byte global -> LDS copy. On gfx1250 use the async-to-LDS DMA path
// (ASYNCcnt-tracked, no VGPR round trip); otherwise plain load/ds_store.
__device__ inline void copy16B_g2l(const _Float16* __restrict__ g,
                                   _Float16* __restrict__ l) {
#ifdef HAVE_ASYNC_LDS
  __builtin_amdgcn_global_load_async_to_lds_b128((gv4i*)(_Float16*)g,
                                                 (lv4i*)l, 0, 0);
#else
  *(float4*)l = *(const float4*)g;
#endif
}

__device__ inline void async_copy_fence() {
#ifdef HAVE_ASYNC_LDS
#ifdef HAVE_WAIT_ASYNC
  __builtin_amdgcn_s_wait_asynccnt(0);
#else
  asm volatile("s_wait_asynccnt 0x0" ::: "memory");
#endif
#endif
}

// ---------------------------------------------------------------------------
// zero-fill float buffer (grid-stride)
// ---------------------------------------------------------------------------
__global__ void zero_kernel(float* __restrict__ p, int n) {
  for (int i = blockIdx.x * blockDim.x + threadIdx.x; i < n;
       i += gridDim.x * blockDim.x)
    p[i] = 0.0f;
}

// ---------------------------------------------------------------------------
// convert f32 weight W[K][N] -> f16 transposed Wt[N][K]
// (so a WMMA B-fragment column is contiguous in memory)
// ---------------------------------------------------------------------------
__global__ void convert_transpose_kernel(const float* __restrict__ W,
                                         _Float16* __restrict__ Wt,
                                         int K, int N) {
  int o = blockIdx.x * blockDim.x + threadIdx.x;
  if (o >= K * N) return;
  int n = o / K;
  int k = o - n * K;
  Wt[o] = (_Float16)W[k * N + n];
}

// ---------------------------------------------------------------------------
// edge scatter: agg[dst] += h[src]; cnt[dst] += 1 (layer 1 only)
// 32 lanes per edge, float4 per lane. Memory/atomic bound; agg is L2-resident.
// ---------------------------------------------------------------------------
__global__ void scatter_kernel(const float* __restrict__ h,
                               const int* __restrict__ edge_index, int E,
                               float* __restrict__ agg, float* __restrict__ cnt,
                               int addCount) {
  int t = blockIdx.x * blockDim.x + threadIdx.x;
  int e = t >> 5;
  int lane = t & 31;
  if (e >= E) return;
  int src = edge_index[e];
  int dst = edge_index[E + e];
  float4 v = *(const float4*)(h + (long long)src * D + lane * 4);
  float* a = agg + (long long)dst * D + lane * 4;
  atomicAdd(a + 0, v.x);
  atomicAdd(a + 1, v.y);
  atomicAdd(a + 2, v.z);
  atomicAdd(a + 3, v.w);
  if (addCount && lane == 0) atomicAdd(cnt + dst, 1.0f);
}

// ---------------------------------------------------------------------------
// WMMA fragment helper (CDNA5 16-bit A/B 16x32 layout):
//   lane L in [0,16):  row/col = L,    K = kb..kb+7 and kb+16..kb+23 (kb=k0)
//   lane L in [16,32): row/col = L-16, same with kb = k0+8
// Fragment = two contiguous 16-byte chunks.
// ---------------------------------------------------------------------------
__device__ inline v16h load_frag(const _Float16* __restrict__ base, int ldk,
                                 int rowc, int kb) {
  v16h f;
  const _Float16* p = base + rowc * ldk + kb;
  ((float4*)&f)[0] = *(const float4*)(p);
  ((float4*)&f)[1] = *(const float4*)(p + 16);
  return f;
}

// ---------------------------------------------------------------------------
// RGCN fused layer: out = relu( (agg/max(cnt,1)) @ Wrel + hin @ Wroot + b )
// block = 256 thr (8 waves). Block owns a 16-row M tile; wave w owns the
// 16-column N tile [16w,16w+16). Weights staged one matrix at a time into
// LDS via async-to-LDS DMA and broadcast to all 8 waves.
// ---------------------------------------------------------------------------
__global__ void __launch_bounds__(256)
rgcn_gemm_kernel(const float* __restrict__ agg, const float* __restrict__ cnt,
                 const float* __restrict__ hin,
                 const _Float16* __restrict__ WrT,    // [128][128] f16 N-major
                 const _Float16* __restrict__ WrootT, // [128][128] f16 N-major
                 const float* __restrict__ bias, float* __restrict__ out,
                 int nrows) {
  __shared__ __align__(16) _Float16 sM[16 * LDK128];
  __shared__ __align__(16) _Float16 sH[16 * LDK128];
  __shared__ __align__(16) _Float16 sW[D * LDK128];  // one weight matrix

  int t = threadIdx.x;
  int grow0 = blockIdx.x * 16;

  // kick off async weight staging (Wrel) -- overlaps with A-tile conversion
  for (int c = t; c < (D * D) / 8; c += 256) {  // 2048 x 16B chunks
    int row = c >> 4;
    int ch = c & 15;
    copy16B_g2l(WrT + row * D + ch * 8, sW + row * LDK128 + ch * 8);
  }

  // stage A tiles (mean-scaled agg, and hin) into LDS as f16
  for (int idx = t; idx < 16 * D; idx += 256) {
    int row = idx >> 7;
    int col = idx & (D - 1);
    int g = grow0 + row;
    float mv = 0.0f, hv = 0.0f;
    if (g < nrows) {
      float c = cnt[g];
      float sc = 1.0f / fmaxf(c, 1.0f);
      mv = agg[(long long)g * D + col] * sc;
      hv = hin[(long long)g * D + col];
    }
    sM[row * LDK128 + col] = (_Float16)mv;
    sH[row * LDK128 + col] = (_Float16)hv;
  }
  async_copy_fence();
  __syncthreads();

  int lane = t & 31;
  int wave = t >> 5;
  int colc = lane & 15;            // local N / local A-row selector
  int col = wave * 16 + colc;      // global N (0..127)
  int krowoff = (lane >> 4) << 3;  // 0 or 8

  v8f acc = {0.f, 0.f, 0.f, 0.f, 0.f, 0.f, 0.f, 0.f};

#pragma unroll
  for (int k0 = 0; k0 < D; k0 += 32) {
    int kb = k0 + krowoff;
    v16h a = load_frag(sM, LDK128, colc, kb);
    v16h b = load_frag(sW, LDK128, col, kb);
    acc = __builtin_amdgcn_wmma_f32_16x16x32_f16(false, a, false, b, (short)0,
                                                 acc, false, false);
  }
  __syncthreads();  // all waves done reading Wrel

  // restage weights: Wroot
  for (int c = t; c < (D * D) / 8; c += 256) {
    int row = c >> 4;
    int ch = c & 15;
    copy16B_g2l(WrootT + row * D + ch * 8, sW + row * LDK128 + ch * 8);
  }
  async_copy_fence();
  __syncthreads();

#pragma unroll
  for (int k0 = 0; k0 < D; k0 += 32) {
    int kb = k0 + krowoff;
    v16h a = load_frag(sH, LDK128, colc, kb);
    v16h b = load_frag(sW, LDK128, col, kb);
    acc = __builtin_amdgcn_wmma_f32_16x16x32_f16(false, a, false, b, (short)0,
                                                 acc, false, false);
  }

  float bv = bias[col];
#pragma unroll
  for (int r = 0; r < 8; ++r) {
    int row = grow0 + r + krowoff;
    if (row < nrows) out[(long long)row * D + col] = fmaxf(acc[r] + bv, 0.0f);
  }
}

// ---------------------------------------------------------------------------
// pair MLP layer 1: z1 = relu( [h[u] || h[v]] @ Wp1 + bp1 )   (K=256, N=128)
// fuses the pair gather + concat into LDS staging; Wp1T staged in two
// K-halves (34 KB each) via async-to-LDS DMA.
// ---------------------------------------------------------------------------
__global__ void __launch_bounds__(256)
pair_mlp1_kernel(const float* __restrict__ h, const int* __restrict__ pairs,
                 const _Float16* __restrict__ Wp1T,  // [128][256] f16 N-major
                 const float* __restrict__ bp1, float* __restrict__ z1,
                 int npairs) {
  __shared__ __align__(16) _Float16 sA[16 * LDK256];
  __shared__ __align__(16) _Float16 sW[D * LDK128];  // one K-half of Wp1T

  int t = threadIdx.x;
  int grow0 = blockIdx.x * 16;

  // kick off staging of K-half 0 of the weights
  for (int c = t; c < (D * D) / 8; c += 256) {
    int row = c >> 4;
    int ch = c & 15;
    copy16B_g2l(Wp1T + row * 256 + ch * 8, sW + row * LDK128 + ch * 8);
  }

  // gather + concat + f32->f16 convert into LDS
  for (int idx = t; idx < 16 * 256; idx += 256) {
    int row = idx >> 8;
    int col = idx & 255;
    int p = grow0 + row;
    float v = 0.0f;
    if (p < npairs) {
      int node = pairs[p * 2 + (col >> 7)];  // col<128 -> u, else v
      v = h[(long long)node * D + (col & (D - 1))];
    }
    sA[row * LDK256 + col] = (_Float16)v;
  }
  async_copy_fence();
  __syncthreads();

  int lane = t & 31;
  int wave = t >> 5;
  int colc = lane & 15;
  int col = wave * 16 + colc;      // N in [0,128)
  int krowoff = (lane >> 4) << 3;

  v8f acc = {0.f, 0.f, 0.f, 0.f, 0.f, 0.f, 0.f, 0.f};

#pragma unroll
  for (int k0 = 0; k0 < 128; k0 += 32) {  // K-half 0
    int kb = k0 + krowoff;
    v16h a = load_frag(sA, LDK256, colc, kb);
    v16h b = load_frag(sW, LDK128, col, kb);
    acc = __builtin_amdgcn_wmma_f32_16x16x32_f16(false, a, false, b, (short)0,
                                                 acc, false, false);
  }
  __syncthreads();

  // stage K-half 1
  for (int c = t; c < (D * D) / 8; c += 256) {
    int row = c >> 4;
    int ch = c & 15;
    copy16B_g2l(Wp1T + row * 256 + 128 + ch * 8, sW + row * LDK128 + ch * 8);
  }
  async_copy_fence();
  __syncthreads();

#pragma unroll
  for (int k0 = 0; k0 < 128; k0 += 32) {  // K-half 1
    int kb = k0 + krowoff;
    v16h a = load_frag(sA, LDK256, colc, 128 + kb);
    v16h b = load_frag(sW, LDK128, col, kb);
    acc = __builtin_amdgcn_wmma_f32_16x16x32_f16(false, a, false, b, (short)0,
                                                 acc, false, false);
  }

  float bv = bp1[col];
#pragma unroll
  for (int r = 0; r < 8; ++r) {
    int row = grow0 + r + krowoff;
    if (row < npairs) z1[(long long)row * D + col] = fmaxf(acc[r] + bv, 0.0f);
  }
}

// ---------------------------------------------------------------------------
// pair MLP layer 2: z2 = relu( z1 @ Wp2 + bp2 )   (K=128, N=64)
// block = 128 thr (4 waves), one 16-col tile per wave. Wp2T (17 KB padded)
// staged via async-to-LDS DMA.
// ---------------------------------------------------------------------------
__global__ void __launch_bounds__(128)
pair_mlp2_kernel(const float* __restrict__ z1,
                 const _Float16* __restrict__ Wp2T,  // [64][128] f16 N-major
                 const float* __restrict__ bp2, float* __restrict__ z2,
                 int npairs) {
  __shared__ __align__(16) _Float16 sA[16 * LDK128];
  __shared__ __align__(16) _Float16 sW[64 * LDK128];

  int t = threadIdx.x;
  int grow0 = blockIdx.x * 16;

  for (int c = t; c < (64 * D) / 8; c += 128) {  // 1024 x 16B chunks
    int row = c >> 4;
    int ch = c & 15;
    copy16B_g2l(Wp2T + row * D + ch * 8, sW + row * LDK128 + ch * 8);
  }

  for (int idx = t; idx < 16 * D; idx += 128) {
    int row = idx >> 7;
    int col = idx & (D - 1);
    int p = grow0 + row;
    float v = (p < npairs) ? z1[(long long)p * D + col] : 0.0f;
    sA[row * LDK128 + col] = (_Float16)v;
  }
  async_copy_fence();
  __syncthreads();

  int lane = t & 31;
  int wave = t >> 5;
  int colc = lane & 15;
  int col = wave * 16 + colc;      // N in [0,64)
  int krowoff = (lane >> 4) << 3;

  v8f acc = {0.f, 0.f, 0.f, 0.f, 0.f, 0.f, 0.f, 0.f};
#pragma unroll
  for (int k0 = 0; k0 < D; k0 += 32) {
    int kb = k0 + krowoff;
    v16h a = load_frag(sA, LDK128, colc, kb);
    v16h b = load_frag(sW, LDK128, col, kb);
    acc = __builtin_amdgcn_wmma_f32_16x16x32_f16(false, a, false, b, (short)0,
                                                 acc, false, false);
  }

  float bv = bp2[col];
#pragma unroll
  for (int r = 0; r < 8; ++r) {
    int row = grow0 + r + krowoff;
    if (row < npairs) z2[(long long)row * 64 + col] = fmaxf(acc[r] + bv, 0.0f);
  }
}

// ---------------------------------------------------------------------------
// head: p = sigmoid( z2 @ Wp3 + bp3 )   (K=64, N=1) -- scalar dot per pair
// (Wp3 is wave-uniform -> scalar loads)
// ---------------------------------------------------------------------------
__global__ void head_kernel(const float* __restrict__ z2,
                            const float* __restrict__ Wp3,
                            const float* __restrict__ bp3,
                            float* __restrict__ out, int npairs) {
  int p = blockIdx.x * blockDim.x + threadIdx.x;
  if (p >= npairs) return;
  const float* row = z2 + (long long)p * 64;
  float acc = bp3[0];
#pragma unroll
  for (int k = 0; k < 64; ++k) acc += row[k] * Wp3[k];
  out[p] = 1.0f / (1.0f + expf(-acc));
}

// ---------------------------------------------------------------------------
// host orchestration
// ---------------------------------------------------------------------------
static inline size_t align256(size_t x) { return (x + 255) & ~(size_t)255; }

extern "C" void kernel_launch(void* const* d_in, const int* in_sizes, int n_in,
                              void* d_out, int out_size, void* d_ws,
                              size_t ws_size, hipStream_t stream) {
  const float* x          = (const float*)d_in[0];
  const int*   edge_index = (const int*)d_in[1];
  // d_in[2] = edge_type (all zero, num_relations==1) -> unused
  const int*   edge_pairs = (const int*)d_in[3];
  const float* Wr1    = (const float*)d_in[4];
  const float* Wroot1 = (const float*)d_in[5];
  const float* b1     = (const float*)d_in[6];
  const float* Wr2    = (const float*)d_in[7];
  const float* Wroot2 = (const float*)d_in[8];
  const float* b2     = (const float*)d_in[9];
  const float* Wp1    = (const float*)d_in[10];
  const float* bp1    = (const float*)d_in[11];
  const float* Wp2    = (const float*)d_in[12];
  const float* bp2    = (const float*)d_in[13];
  const float* Wp3    = (const float*)d_in[14];
  const float* bp3    = (const float*)d_in[15];
  float* out = (float*)d_out;

  const int N = in_sizes[0] / D;       // 100000 nodes
  const int E = in_sizes[1] / 2;       // 600000 edges
  const int P = in_sizes[3] / 2;       // 100000 pairs

  // -------- workspace carve-up (all offsets 256B aligned) --------
  char* ws = (char*)d_ws;
  size_t off = 0;
  float* agg = (float*)(ws + off); off = align256(off + (size_t)N * D * 4);
  float* h1  = (float*)(ws + off); off = align256(off + (size_t)N * D * 4);
  float* h2  = (float*)(ws + off); off = align256(off + (size_t)N * D * 4);
  float* cnt = (float*)(ws + off); off = align256(off + (size_t)N * 4);
  _Float16* Wr1T    = (_Float16*)(ws + off); off = align256(off + (size_t)D * D * 2);
  _Float16* Wroot1T = (_Float16*)(ws + off); off = align256(off + (size_t)D * D * 2);
  _Float16* Wr2T    = (_Float16*)(ws + off); off = align256(off + (size_t)D * D * 2);
  _Float16* Wroot2T = (_Float16*)(ws + off); off = align256(off + (size_t)D * D * 2);
  _Float16* Wp1T    = (_Float16*)(ws + off); off = align256(off + (size_t)2 * D * D * 2);
  _Float16* Wp2T    = (_Float16*)(ws + off); off = align256(off + (size_t)D * 64 * 2);
  float* z1 = agg;  // free after layer-2 GEMM
  float* z2 = h1;   // free after layer-2 GEMM
  (void)ws_size; (void)n_in; (void)out_size;

  const int ZB = 2048;

  // -------- weight precondition: f32 -> f16 transposed --------
  {
    int n;
    n = D * D;
    convert_transpose_kernel<<<(n + 255) / 256, 256, 0, stream>>>(Wr1, Wr1T, D, D);
    convert_transpose_kernel<<<(n + 255) / 256, 256, 0, stream>>>(Wroot1, Wroot1T, D, D);
    convert_transpose_kernel<<<(n + 255) / 256, 256, 0, stream>>>(Wr2, Wr2T, D, D);
    convert_transpose_kernel<<<(n + 255) / 256, 256, 0, stream>>>(Wroot2, Wroot2T, D, D);
    n = 2 * D * D;
    convert_transpose_kernel<<<(n + 255) / 256, 256, 0, stream>>>(Wp1, Wp1T, 2 * D, D);
    n = D * 64;
    convert_transpose_kernel<<<(n + 255) / 256, 256, 0, stream>>>(Wp2, Wp2T, D, 64);
  }

  // -------- layer 1 --------
  zero_kernel<<<ZB, 256, 0, stream>>>(agg, N * D);
  zero_kernel<<<ZB, 256, 0, stream>>>(cnt, N);
  {
    long long tthr = (long long)E * 32;
    scatter_kernel<<<(int)((tthr + 255) / 256), 256, 0, stream>>>(
        x, edge_index, E, agg, cnt, /*addCount=*/1);
  }
  rgcn_gemm_kernel<<<(N + 15) / 16, 256, 0, stream>>>(agg, cnt, x, Wr1T,
                                                      Wroot1T, b1, h1, N);

  // -------- layer 2 (degree counts reused) --------
  zero_kernel<<<ZB, 256, 0, stream>>>(agg, N * D);
  {
    long long tthr = (long long)E * 32;
    scatter_kernel<<<(int)((tthr + 255) / 256), 256, 0, stream>>>(
        h1, edge_index, E, agg, cnt, /*addCount=*/0);
  }
  rgcn_gemm_kernel<<<(N + 15) / 16, 256, 0, stream>>>(agg, cnt, h1, Wr2T,
                                                      Wroot2T, b2, h2, N);

  // -------- link-predictor MLP (gather fused into layer 1) --------
  pair_mlp1_kernel<<<(P + 15) / 16, 256, 0, stream>>>(h2, edge_pairs, Wp1T,
                                                      bp1, z1, P);
  pair_mlp2_kernel<<<(P + 15) / 16, 128, 0, stream>>>(z1, Wp2T, bp2, z2, P);
  head_kernel<<<(P + 255) / 256, 256, 0, stream>>>(z2, Wp3, bp3, out, P);
}
